// TransformerBlock_515396076425
// MI455X (gfx1250) — compile-verified, matrix operand order checked by reading, not run on hardware
//
#include <hip/hip_runtime.h>
#include <hip/hip_bf16.h>
#include <math.h>

// ---------------------------------------------------------------------------
// MI455X / gfx1250: wave32, WMMA 16x16x32 bf16 with f32 accumulate.
// ---------------------------------------------------------------------------

typedef __attribute__((ext_vector_type(16))) __bf16 v16bf;
typedef __attribute__((ext_vector_type(8)))  float  v8f;

#define DEVINL __device__ __forceinline__

DEVINL uint16_t f2bf(float f) {
    uint32_t u = __float_as_uint(f);
    uint32_t r = (u + 0x7FFFu + ((u >> 16) & 1u)) >> 16;  // RNE
    return (uint16_t)r;
}
DEVINL float bf2f(uint16_t u) { return __uint_as_float(((uint32_t)u) << 16); }
DEVINL float gelu_f(float x) { return 0.5f * x * (1.0f + erff(x * 0.70710678118654752f)); }

union FragBF { uint4 q[2]; v16bf v; };
union FragF  { v8f v; float f[8]; };

// A-fragment (16x32 bf16, row-major source, stride ld): per ISA layout,
// lane l holds row m=l&15; halves 0..7 at K=8*(l>>4)+0..7, halves 8..15 at
// K=16+8*(l>>4)+0..7 -> two contiguous 16B chunks per lane.
// B-fragment mirrors this, loaded from B^T stored [N,K] row-major.
DEVINL v16bf load_frag(const uint16_t* base, int ld, int lane) {
    const int r = lane & 15, h = lane >> 4;
    const uint16_t* p = base + (size_t)r * ld + 8 * h;
    FragBF fr;
    fr.q[0] = *reinterpret_cast<const uint4*>(p);
    fr.q[1] = *reinterpret_cast<const uint4*>(p + 16);
    return fr.v;
}

// ---------------------------------------------------------------------------
// Generic batched TN GEMM:  C[m,n] = alpha * sum_k A[m,k] * BT[n,k]  (+epilogue)
// block = 256 threads = 8 waves; wave w owns n-tile (blockIdx.y*8+w).
// Batch offsets: off = (z/zdiv)*so + (z%zdiv)*si  (handles (g,t)-style batching)
// ---------------------------------------------------------------------------
enum { EPI_F32 = 0, EPI_F32_ACC = 1, EPI_BF16 = 2, EPI_BIAS_GELU_BF16 = 3, EPI_BIAS_RES_F32 = 4 };

template <int EPI>
__global__ __launch_bounds__(256)
void gemm_bf16_tn(const uint16_t* __restrict__ A, const uint16_t* __restrict__ BT,
                  float* __restrict__ Cf, uint16_t* __restrict__ Cb,
                  const float* __restrict__ bias, const float* __restrict__ res,
                  int K, int lda, int ldb, int ldc, int ldres, float alpha,
                  int zdiv, long aso, long asi, long bso, long bsi, long cso, long csi)
{
    const int z  = blockIdx.z;
    const int zo = z / zdiv, zi = z - zo * zdiv;
    A  += (long)zo * aso + (long)zi * asi;
    BT += (long)zo * bso + (long)zi * bsi;
    const long coff = (long)zo * cso + (long)zi * csi;

    const int lane = threadIdx.x & 31;
    const int wave = threadIdx.x >> 5;
    const int m0   = blockIdx.x * 16;
    const int n0   = (blockIdx.y * 8 + wave) * 16;

    FragF acc;
#pragma unroll
    for (int i = 0; i < 8; ++i) acc.f[i] = 0.0f;
    v8f c = acc.v;

    for (int k0 = 0; k0 < K; k0 += 32) {
        v16bf a = load_frag(A  + (size_t)m0 * lda + k0, lda, lane);
        v16bf b = load_frag(BT + (size_t)n0 * ldb + k0, ldb, lane);
        c = __builtin_amdgcn_wmma_f32_16x16x32_bf16(false, a, false, b, (short)0, c, false, false);
    }
    acc.v = c;

    // C/D layout: lane -> n = n0 + (lane&15); vgpr i -> m = m0 + 8*(lane>>4) + i
    const int nn = n0 + (lane & 15);
    const int mh = m0 + 8 * (lane >> 4);
#pragma unroll
    for (int i = 0; i < 8; ++i) {
        const int  m   = mh + i;
        const long idx = coff + (long)m * ldc + nn;
        float v = acc.f[i] * alpha;
        if constexpr (EPI == EPI_F32)               Cf[idx] = v;
        else if constexpr (EPI == EPI_F32_ACC)      Cf[idx] += v;
        else if constexpr (EPI == EPI_BF16)         Cb[idx] = f2bf(v);
        else if constexpr (EPI == EPI_BIAS_GELU_BF16) Cb[idx] = f2bf(gelu_f(v + bias[nn]));
        else /* EPI_BIAS_RES_F32 */                 Cf[idx] = v + bias[nn] + res[(long)m * ldres + nn];
    }
}

// ---------------------------------------------------------------------------
// Small helper kernels
// ---------------------------------------------------------------------------
__global__ void f32_to_bf16(const float* __restrict__ in, uint16_t* __restrict__ out, int n) {
    int i = blockIdx.x * blockDim.x + threadIdx.x;
    if (i < n) out[i] = f2bf(in[i]);
}

__global__ void fill_zero_b16(uint16_t* __restrict__ p, int n) {
    int i = blockIdx.x * blockDim.x + threadIdx.x;
    if (i < n) p[i] = 0;
}

// w0p[g,t,u,ci] and w0pt[g,t,ci,u] bf16 repacks of conv1_w (WIN*C, C, 3, 3)
__global__ void pack_w0(const float* __restrict__ w, uint16_t* __restrict__ w0p,
                        uint16_t* __restrict__ w0pt) {
    int i = blockIdx.x * 256 + threadIdx.x;
    if (i >= 4 * 9 * 128 * 128) return;
    int ci = i & 127;
    int u  = (i >> 7) & 127;
    int gt = i >> 14;
    int t  = gt % 9;
    int g  = gt / 9;
    uint16_t b = f2bf(w[(((g * 128 + u) * 128 + ci) * 9) + t]);
    w0p [((g * 9 + t) * 128 + u ) * 128 + ci] = b;
    w0pt[((g * 9 + t) * 128 + ci) * 128 + u ] = b;
}

// wfsum[o2] = sum_u w_fus[o2, 512+u]  (for the b_gk bias folded into Keff)
__global__ void wfsum_kernel(const float* __restrict__ w_fus, float* __restrict__ wfsum) {
    int o = threadIdx.x;
    float s = 0.0f;
    for (int u = 0; u < 128; ++u) s += w_fus[o * 640 + 512 + u];
    wfsum[o] = s;
}

// LayerNorm over C=128 (one block of 128 threads per row), bf16 output
__global__ __launch_bounds__(128)
void ln_bf16(const float* __restrict__ in, const float* __restrict__ gm,
             const float* __restrict__ bt, uint16_t* __restrict__ out) {
    const int row = blockIdx.x;
    const int tid = threadIdx.x;
    __shared__ float red[128];
    float v = in[(long)row * 128 + tid];
    red[tid] = v; __syncthreads();
    for (int s = 64; s > 0; s >>= 1) { if (tid < s) red[tid] += red[tid + s]; __syncthreads(); }
    float mean = red[0] * (1.0f / 128.0f);
    __syncthreads();
    float d = v - mean;
    red[tid] = d * d; __syncthreads();
    for (int s = 64; s > 0; s >>= 1) { if (tid < s) red[tid] += red[tid + s]; __syncthreads(); }
    float var = red[0] * (1.0f / 128.0f);
    out[(long)row * 128 + tid] = f2bf(d * rsqrtf(var + 1e-5f) * gm[tid] + bt[tid]);
}

// Window partial sums -> T[b,g,ci,t] (mean-of-conv trick). One wave per (b,g,ci),
// lane = window row. wave32 shuffles.
__global__ __launch_bounds__(256)
void winsums(const uint16_t* __restrict__ xnb, uint16_t* __restrict__ Tpad) {
    const int lane = threadIdx.x & 31;
    const int item = blockIdx.x * 8 + (threadIdx.x >> 5);  // 8*4*128 = 4096 items
    const int ci = item & 127;
    const int g  = (item >> 7) & 3;
    const int b  = item >> 9;
    const int gh = g >> 1, gw = g & 1;
    const long base = (((long)b * 64 + gh * 32) * 64 + gw * 32) * 128 + ci;

    float rs = 0.0f, v0 = 0.0f, v31 = 0.0f;
    for (int cc = 0; cc < 32; ++cc) {
        float x = bf2f(xnb[base + ((long)lane * 64 + cc) * 128]);
        rs += x;
        if (cc == 0)  v0  = x;
        if (cc == 31) v31 = x;
    }
    float S = rs, C0 = v0, C31 = v31;
    for (int m = 16; m > 0; m >>= 1) {
        S   += __shfl_xor(S, m);
        C0  += __shfl_xor(C0, m);
        C31 += __shfl_xor(C31, m);
    }
    float R0 = __shfl(rs, 0),  R31 = __shfl(rs, 31);
    float c00  = __shfl(v0, 0),  c310  = __shfl(v0, 31);
    float c031 = __shfl(v31, 0), c3131 = __shfl(v31, 31);

    if (lane < 9) {
        int kh = lane / 3, kw = lane % 3;
        int dh = kh - 1, dw = kw - 1;
        float rowEx = (dh == 1) ? R0 : (dh == -1) ? R31 : 0.0f;
        float colEx = (dw == 1) ? C0 : (dw == -1) ? C31 : 0.0f;
        float corner = 0.0f;
        if (dh != 0 && dw != 0) {
            if (dh == 1  && dw == 1 ) corner = c00;    // x[0,0]
            if (dh == 1  && dw == -1) corner = c031;   // x[0,31]
            if (dh == -1 && dw == 1 ) corner = c310;   // x[31,0]
            if (dh == -1 && dw == -1) corner = c3131;  // x[31,31]
        }
        float T = S - rowEx - colEx + corner;
        Tpad[((long)g * 16 + b) * 1152 + ci * 9 + lane] = f2bf(T);
    }
}

// SE over windows: pooled[b,g,:] -> wgt[b,g]. Tiny, one block.
__global__ __launch_bounds__(128)
void se_mlp(const float* __restrict__ pooled, const float* __restrict__ wd,
            const float* __restrict__ bd, const float* __restrict__ w1,
            const float* __restrict__ b1, const float* __restrict__ w2,
            const float* __restrict__ b2, float* __restrict__ wgt) {
    __shared__ float ly[32];
    __shared__ float lh[128];
    const int tid = threadIdx.x;
    if (tid < 32) {
        int b = tid >> 2, g = tid & 3;
        float s = bd[g];
        for (int ci = 0; ci < 128; ++ci)
            s += pooled[((long)g * 16 + b) * 128 + ci] * wd[g * 128 + ci];
        ly[tid] = s;
    }
    __syncthreads();
    {
        int b = tid >> 4, i = tid & 15;
        float s = b1[i];
        for (int g = 0; g < 4; ++g) s += ly[b * 4 + g] * w1[i * 4 + g];
        lh[tid] = gelu_f(s);
    }
    __syncthreads();
    if (tid < 32) {
        int b = tid >> 2, g = tid & 3;
        float s = b2[g];
        for (int i = 0; i < 16; ++i) s += lh[b * 16 + i] * w2[g * 16 + i];
        wgt[tid] = 1.0f / (1.0f + expf(-s));
    }
}

// Keff[b][o2][t*128+ci] = sum_g wgt[b,g]*P[g,t][o2,ci] + b_gk[ci]*wfsum[o2]
__global__ __launch_bounds__(256)
void build_keff(const float* __restrict__ P, const float* __restrict__ wgt,
                const float* __restrict__ bgk, const float* __restrict__ wfsum,
                uint16_t* __restrict__ keff) {
    int i = blockIdx.x * 256 + threadIdx.x;   // [b][o2][t][ci]
    if (i >= 8 * 128 * 1152) return;
    int ci = i & 127;
    int j  = i >> 7;
    int t  = j % 9;
    int oj = j / 9;
    int o2 = oj & 127;
    int b  = oj >> 7;
    float v = bgk[ci] * wfsum[o2];
#pragma unroll
    for (int g = 0; g < 4; ++g)
        v += wgt[b * 4 + g] * P[(((long)(g * 9 + t)) * 128 + o2) * 128 + ci];
    keff[i] = f2bf(v);
}

// Effective dynamic conv: implicit-GEMM 3x3, C->C over 64x64, per batch.
// Block = one image row (64 positions) x all 128 out channels. LDS stages
// 3 halo rows (3 x 66 x 128 bf16 = 50.7 KB). 36 WMMAs per 16x16 tile.
// Epilogue: + b_fus[o2] + residual x -> y1 (fp32).
__global__ __launch_bounds__(256)
void dyn_conv(const uint16_t* __restrict__ xnb, const uint16_t* __restrict__ keff,
              const float* __restrict__ bfus, const float* __restrict__ xres,
              float* __restrict__ y1) {
    const int y = blockIdx.x;    // 0..63
    const int b = blockIdx.y;    // 0..7
    const int tid  = threadIdx.x;
    const int lane = tid & 31, wave = tid >> 5;
    __shared__ __align__(16) uint16_t sh[3 * 66 * 128];
    const uint4 z4 = make_uint4(0u, 0u, 0u, 0u);

    for (int rr = 0; rr < 3; ++rr) {
        const int yy = y + rr - 1;
        const bool ok = (yy >= 0) && (yy < 64);
        for (int i = tid; i < 1024; i += 256) {   // 64 cols x 128 ch = 1024 uint4
            uint4 v = z4;
            if (ok) {
                const uint16_t* src = xnb + (((long)b * 64 + yy) * 64) * 128;
                v = reinterpret_cast<const uint4*>(src)[i];
            }
            *reinterpret_cast<uint4*>(&sh[(rr * 66 + 1) * 128 + i * 8]) = v;
        }
        for (int i = tid; i < 32; i += 256) {     // zero halo cols 0 and 65
            int col = (i < 16) ? 0 : 65;
            int jj  = i & 15;
            *reinterpret_cast<uint4*>(&sh[(rr * 66 + col) * 128 + jj * 8]) = z4;
        }
    }
    __syncthreads();

    const uint16_t* kb = keff + (long)b * 128 * 1152;
    const int n0  = wave * 16;
    const int r15 = lane & 15, hh = lane >> 4;

    for (int mt = 0; mt < 4; ++mt) {
        const int m0 = mt * 16;
        FragF acc;
#pragma unroll
        for (int i = 0; i < 8; ++i) acc.f[i] = 0.0f;
        v8f c = acc.v;
#pragma unroll
        for (int t = 0; t < 9; ++t) {
            const int kh = t / 3, kw = t % 3;
            const uint16_t* arow = &sh[((kh * 66) + m0 + r15 + kw) * 128 + 8 * hh];
            const uint16_t* brow = kb + (long)(n0 + r15) * 1152 + t * 128 + 8 * hh;
#pragma unroll
            for (int c0 = 0; c0 < 128; c0 += 32) {
                FragBF fa, fb;
                fa.q[0] = *reinterpret_cast<const uint4*>(arow + c0);
                fa.q[1] = *reinterpret_cast<const uint4*>(arow + c0 + 16);
                fb.q[0] = *reinterpret_cast<const uint4*>(brow + c0);
                fb.q[1] = *reinterpret_cast<const uint4*>(brow + c0 + 16);
                c = __builtin_amdgcn_wmma_f32_16x16x32_bf16(false, fa.v, false, fb.v,
                                                            (short)0, c, false, false);
            }
        }
        acc.v = c;
        const int nn = n0 + r15;
#pragma unroll
        for (int i = 0; i < 8; ++i) {
            const int  x = m0 + 8 * hh + i;
            const long p = ((long)b * 4096 + y * 64 + x) * 128 + nn;
            y1[p] = acc.f[i] + bfus[nn] + xres[p];
        }
    }
}

// Depthwise 3x3 over HID=512, NHWC bf16, fused bias + GELU. Memory-bound; VALU.
__global__ __launch_bounds__(256)
void dw_conv(const uint16_t* __restrict__ hin, const float* __restrict__ w,
             const float* __restrict__ bias, uint16_t* __restrict__ hout) {
    long i = (long)blockIdx.x * 256 + threadIdx.x;
    if (i >= (long)8 * 64 * 64 * 512) return;
    int c  = (int)(i & 511);
    int x  = (int)((i >> 9) & 63);
    int yv = (int)((i >> 15) & 63);
    int b  = (int)(i >> 21);
    float acc = bias[c];
#pragma unroll
    for (int kh = 0; kh < 3; ++kh) {
        int yy = yv + kh - 1;
        if (yy < 0 || yy > 63) continue;
#pragma unroll
        for (int kw = 0; kw < 3; ++kw) {
            int xx = x + kw - 1;
            if (xx < 0 || xx > 63) continue;
            acc += bf2f(hin[((((long)b * 64 + yy) * 64 + xx) << 9) + c]) * w[c * 9 + kh * 3 + kw];
        }
    }
    hout[i] = f2bf(gelu_f(acc));
}

// ---------------------------------------------------------------------------
// Host launcher
// ---------------------------------------------------------------------------
extern "C" void kernel_launch(void* const* d_in, const int* in_sizes, int n_in,
                              void* d_out, int out_size, void* d_ws, size_t ws_size,
                              hipStream_t stream) {
    (void)in_sizes; (void)n_in; (void)out_size; (void)ws_size;
    const float* x      = (const float*)d_in[0];
    const float* ln1_g  = (const float*)d_in[1];
    const float* ln1_b  = (const float*)d_in[2];
    const float* conv1w = (const float*)d_in[3];
    const float* w_down = (const float*)d_in[4];
    const float* b_down = (const float*)d_in[5];
    const float* w1     = (const float*)d_in[6];
    const float* b1     = (const float*)d_in[7];
    const float* w2     = (const float*)d_in[8];
    const float* b2     = (const float*)d_in[9];
    const float* w_gk   = (const float*)d_in[10];
    const float* b_gk   = (const float*)d_in[11];
    const float* w_fus  = (const float*)d_in[12];
    const float* b_fus  = (const float*)d_in[13];
    const float* ln2_g  = (const float*)d_in[14];
    const float* ln2_b  = (const float*)d_in[15];
    const float* lin1_w = (const float*)d_in[16];
    const float* lin1_b = (const float*)d_in[17];
    const float* dw_w   = (const float*)d_in[18];
    const float* dw_b   = (const float*)d_in[19];
    const float* lin2_w = (const float*)d_in[20];
    const float* lin2_b = (const float*)d_in[21];
    float* out = (float*)d_out;

    char* ws = (char*)d_ws;
    size_t off = 0;
    auto alloc = [&](size_t bytes) -> char* {
        size_t o = (off + 255) & ~(size_t)255;
        off = o + bytes;
        return ws + o;
    };
    uint16_t* w1b     = (uint16_t*)alloc((size_t)589824 * 2);   // conv1_w bf16 [(g,co), ci*9+t]
    uint16_t* w_gkb   = (uint16_t*)alloc((size_t)65536 * 2);
    uint16_t* w_fusb  = (uint16_t*)alloc((size_t)81920 * 2);
    uint16_t* lin1_wb = (uint16_t*)alloc((size_t)65536 * 2);
    uint16_t* lin2_wb = (uint16_t*)alloc((size_t)65536 * 2);
    uint16_t* w0p     = (uint16_t*)alloc((size_t)589824 * 2);   // [g,t,u,ci]
    uint16_t* w0pt    = (uint16_t*)alloc((size_t)589824 * 2);   // [g,t,ci,u]
    uint16_t* Mtb     = (uint16_t*)alloc((size_t)589824 * 2);   // [g][o][t*128+u]
    float*    P       = (float*)   alloc((size_t)36 * 16384 * 4);
    float*    wfsum   = (float*)   alloc((size_t)128 * 4);
    uint16_t* xnb     = (uint16_t*)alloc((size_t)4194304 * 2);  // LN1(x) bf16 NHWC
    uint16_t* Tpad    = (uint16_t*)alloc((size_t)73728 * 2);    // [g][16][1152]
    float*    pooled  = (float*)   alloc((size_t)8192 * 4);     // [g][16][128]
    float*    wgt     = (float*)   alloc((size_t)32 * 4);
    uint16_t* keffb   = (uint16_t*)alloc((size_t)1179648 * 2);  // [b][o2][t*128+ci]
    float*    y1      = (float*)   alloc((size_t)4194304 * 4);  // after attn branch
    uint16_t* h2b     = (uint16_t*)alloc((size_t)4194304 * 2);  // LN2 bf16
    uint16_t* hh      = (uint16_t*)alloc((size_t)16777216 * 2); // lin1 out
    uint16_t* hd      = (uint16_t*)alloc((size_t)16777216 * 2); // dwconv out

    // ---- batch-independent weight preprocessing -------------------------
    f32_to_bf16<<<2304, 256, 0, stream>>>(conv1w, w1b, 589824);
    f32_to_bf16<<<256,  256, 0, stream>>>(w_gk,   w_gkb, 65536);
    f32_to_bf16<<<320,  256, 0, stream>>>(w_fus,  w_fusb, 81920);
    f32_to_bf16<<<256,  256, 0, stream>>>(lin1_w, lin1_wb, 65536);
    f32_to_bf16<<<256,  256, 0, stream>>>(lin2_w, lin2_wb, 65536);
    pack_w0<<<2304, 256, 0, stream>>>(conv1w, w0p, w0pt);
    wfsum_kernel<<<1, 128, 0, stream>>>(w_fus, wfsum);

    // Mt[g][o][(t,u)] = sum_ci w_gk[o, g*128+ci] * w0[g,u,ci,t]   (bf16 out)
    gemm_bf16_tn<EPI_BF16><<<dim3(8, 9, 4), 256, 0, stream>>>(
        w_gkb, w0p, nullptr, Mtb, nullptr, nullptr,
        /*K*/128, /*lda*/512, /*ldb*/128, /*ldc*/1152, 0, 1.0f,
        /*zdiv*/1, /*aso*/128, 0, /*bso*/147456, 0, /*cso*/147456, 0);

    // P[g,t][o2,ci] = w_fus_g x w0pt   (A1)
    gemm_bf16_tn<EPI_F32><<<dim3(8, 1, 36), 256, 0, stream>>>(
        w_fusb, w0pt, P, nullptr, nullptr, nullptr,
        128, 640, 128, 128, 0, 1.0f,
        /*zdiv*/9, /*aso*/128, 0, /*bso*/147456, /*bsi*/16384, /*cso*/147456, /*csi*/16384);

    // P += w_fus_4 x Mt   (A2, accumulate)
    gemm_bf16_tn<EPI_F32_ACC><<<dim3(8, 1, 36), 256, 0, stream>>>(
        w_fusb + 512, Mtb, P, nullptr, nullptr, nullptr,
        128, 640, 1152, 128, 0, 1.0f,
        9, 0, 0, /*bso*/147456, /*bsi*/128, 147456, 16384);

    // ---- main pipeline --------------------------------------------------
    ln_bf16<<<32768, 128, 0, stream>>>(x, ln1_g, ln1_b, xnb);

    fill_zero_b16<<<288, 256, 0, stream>>>(Tpad, 73728);
    winsums<<<512, 256, 0, stream>>>(xnb, Tpad);

    // pooled[g][b][co] = (1/1024) * Tpad x conv1_w^T
    gemm_bf16_tn<EPI_F32><<<dim3(1, 1, 4), 256, 0, stream>>>(
        Tpad, w1b, pooled, nullptr, nullptr, nullptr,
        1152, 1152, 1152, 128, 0, 1.0f / 1024.0f,
        1, /*aso*/18432, 0, /*bso*/147456, 0, /*cso*/2048, 0);

    se_mlp<<<1, 128, 0, stream>>>(pooled, w_down, b_down, w1, b1, w2, b2, wgt);

    build_keff<<<4608, 256, 0, stream>>>(P, wgt, b_gk, wfsum, keffb);

    dyn_conv<<<dim3(64, 8), 256, 0, stream>>>(xnb, keffb, b_fus, x, y1);

    ln_bf16<<<32768, 128, 0, stream>>>(y1, ln2_g, ln2_b, h2b);

    // lin1: [32768,128] x [512,128]^T, bias+GELU -> bf16
    gemm_bf16_tn<EPI_BIAS_GELU_BF16><<<dim3(2048, 4, 1), 256, 0, stream>>>(
        h2b, lin1_wb, nullptr, hh, lin1_b, nullptr,
        128, 128, 128, 512, 0, 1.0f,
        1, 0, 0, 0, 0, 0, 0);

    dw_conv<<<65536, 256, 0, stream>>>(hh, dw_w, dw_b, hd);

    // lin2: [32768,512] x [128,512]^T, bias + residual(y1) -> fp32 out
    gemm_bf16_tn<EPI_BIAS_RES_F32><<<dim3(2048, 1, 1), 256, 0, stream>>>(
        hd, lin2_wb, out, nullptr, lin2_b, y1,
        512, 512, 512, 128, /*ldres*/128, 1.0f,
        1, 0, 0, 0, 0, 0, 0);
}
